// MultiHeadedAttention_22634477650437
// MI455X (gfx1250) — compile-verified, matrix operand order checked by reading, not run on hardware
//
#include <hip/hip_runtime.h>
#include <hip/hip_bf16.h>
#include <stdint.h>

#define D_MODEL 1024
#define NH      16
#define DH      64
#define SEQ     2048
#define BATCH   2
#define NEG_BIG (-3.0e38f)
#define MWPR    (SEQ / 32)          /* mask words per row = 64 */

typedef __attribute__((ext_vector_type(16))) __bf16 v16bf;
typedef __attribute__((ext_vector_type(8)))  float  v8f;

union AFrag {
    v16bf  v;
    __bf16 h[16];
    uint4  q[2];
};

// Load 8 contiguous bf16 (16 bytes) into fragment slots.
__device__ __forceinline__ void ld8_bf16(const __bf16* p, __bf16* dst) {
    *(uint4*)dst = *(const uint4*)p;
}

// ---------------------------------------------------------------------------
// fp32 -> bf16 bulk conversion (8 elements / thread, b128 in, b128 out).
// Amortized: weights are re-read by 64 m-tiles, activations by 16 n-tiles.
// ---------------------------------------------------------------------------
__global__ __launch_bounds__(256) void cvt_bf16(const float* __restrict__ s,
                                                __bf16* __restrict__ d, int n8) {
    const int i = blockIdx.x * 256 + threadIdx.x;
    if (i >= n8) return;
    const float4* p = (const float4*)s + (size_t)i * 2;
    float4 a = p[0];
    float4 b = p[1];
    union { uint4 q; __bf16 h[8]; } pk;
    pk.h[0] = (__bf16)a.x; pk.h[1] = (__bf16)a.y; pk.h[2] = (__bf16)a.z; pk.h[3] = (__bf16)a.w;
    pk.h[4] = (__bf16)b.x; pk.h[5] = (__bf16)b.y; pk.h[6] = (__bf16)b.z; pk.h[7] = (__bf16)b.w;
    ((uint4*)d)[i] = pk.q;
}

// ---------------------------------------------------------------------------
// Pack bool mask bytes into bitmask words via wave ballot (8.4MB -> 1MB).
// ---------------------------------------------------------------------------
__global__ __launch_bounds__(256) void maskpack(const uint8_t* __restrict__ m,
                                                uint32_t* __restrict__ w, int n) {
    const int i = blockIdx.x * 256 + threadIdx.x;
    if (i >= n) return;
    unsigned long long bal = __ballot(m[i] != 0);
    if ((threadIdx.x & 31) == 0) w[i >> 5] = (uint32_t)bal;
}

// ---------------------------------------------------------------------------
// All-bf16 GEMM: C[M,1024] = A[M,1024] x W[1024,1024]^T + bias
// One wave computes a 64(M) x 64(N) tile: 16 accumulators, K step 32.
// Per K step: 16 x global_load_b128 feeding 16 x v_wmma (1 load / WMMA).
// MODE 0: C bf16 row-major          (Q, K projections)
// MODE 1: C bf16 transposed [N,M]   (V projection -> contiguous WMMA B-frags)
// MODE 2: C fp32 row-major          (output projection into d_out)
// ---------------------------------------------------------------------------
template <int MODE>
__global__ __launch_bounds__(256) void gemm64(const __bf16* __restrict__ Ab,
                                              const __bf16* __restrict__ Wb,
                                              const float* __restrict__ bias,
                                              void* __restrict__ outp, int M) {
    const int lane = threadIdx.x & 31;
    const int wave = threadIdx.x >> 5;
    const int ln   = lane & 15;
    const int half = lane >> 4;
    const int wg   = blockIdx.x * 8 + wave;
    const int NT   = D_MODEL / 64;                 // 16 n-tiles of 64
    const int tm   = wg / NT;
    const int tn   = wg % NT;
    if (tm * 64 >= M) return;

    const v8f z = {0.f, 0.f, 0.f, 0.f, 0.f, 0.f, 0.f, 0.f};
    v8f acc[4][4];
#pragma unroll
    for (int ms = 0; ms < 4; ++ms)
#pragma unroll
        for (int nt = 0; nt < 4; ++nt) acc[ms][nt] = z;

    const __bf16* arow[4];
    const __bf16* brow[4];
#pragma unroll
    for (int t = 0; t < 4; ++t) {
        arow[t] = Ab + (size_t)(tm * 64 + t * 16 + ln) * D_MODEL;
        brow[t] = Wb + (size_t)(tn * 64 + t * 16 + ln) * D_MODEL;
    }

    for (int k0 = 0; k0 < D_MODEL; k0 += 32) {
        AFrag a[4], bf[4];
#pragma unroll
        for (int t = 0; t < 4; ++t) {
            ld8_bf16(arow[t] + k0 + half * 8,      a[t].h);
            ld8_bf16(arow[t] + k0 + 16 + half * 8, a[t].h + 8);
            ld8_bf16(brow[t] + k0 + half * 8,      bf[t].h);
            ld8_bf16(brow[t] + k0 + 16 + half * 8, bf[t].h + 8);
        }
#pragma unroll
        for (int ms = 0; ms < 4; ++ms)
#pragma unroll
            for (int nt = 0; nt < 4; ++nt)
                acc[ms][nt] = __builtin_amdgcn_wmma_f32_16x16x32_bf16(
                    false, a[ms].v, false, bf[nt].v, (short)0, acc[ms][nt],
                    false, false);
    }

#pragma unroll
    for (int ms = 0; ms < 4; ++ms) {
#pragma unroll
        for (int nt = 0; nt < 4; ++nt) {
            const int n = tn * 64 + nt * 16 + ln;
            const float bv = bias[n];
            if (MODE == 0) {
                __bf16* ob = (__bf16*)outp;
#pragma unroll
                for (int r = 0; r < 8; ++r) {
                    const int m = tm * 64 + ms * 16 + half * 8 + r;
                    ob[(size_t)m * D_MODEL + n] = (__bf16)(acc[ms][nt][r] + bv);
                }
            } else if (MODE == 1) {
                // VT[b, n, s]: rows m..m+7 contiguous in s -> one 16B store.
                __bf16* ob = (__bf16*)outp;
                const int row0 = tm * 64 + ms * 16 + half * 8;
                const int bidx = row0 / SEQ;
                const int s0   = row0 % SEQ;
                union { uint4 q; __bf16 h[8]; } pk;
#pragma unroll
                for (int r = 0; r < 8; ++r) pk.h[r] = (__bf16)(acc[ms][nt][r] + bv);
                *(uint4*)(ob + ((size_t)(bidx * D_MODEL + n) * SEQ + s0)) = pk.q;
            } else {
                float* of = (float*)outp;
#pragma unroll
                for (int r = 0; r < 8; ++r) {
                    const int m = tm * 64 + ms * 16 + half * 8 + r;
                    of[(size_t)m * D_MODEL + n] = acc[ms][nt][r] + bv;
                }
            }
        }
    }
}

// ---------------------------------------------------------------------------
// Flash attention: one wave per (b, h, 16-query tile), kv tiles of 32.
// Scores C-layout (m = r + 8*(lane>=16), n = lane%16); shfl_xor row
// reductions; P bounced through per-wave LDS into A-fragment layout.
// Mask words are streamed ahead with the CDNA5 async-to-LDS engine
// (double-buffered global_load_async_to_lds_b32 + s_wait_asynccnt).
// ---------------------------------------------------------------------------
__global__ __launch_bounds__(256) void attn16(const __bf16* __restrict__ qb,
                                              const __bf16* __restrict__ kb,
                                              const __bf16* __restrict__ vtb,
                                              const uint32_t* __restrict__ mw,
                                              __bf16* __restrict__ ob) {
    __shared__ __align__(16) __bf16  plds[8][16][32];   // P bounce, 1KB/wave
    __shared__ __align__(16) uint32_t mlds[8][2][16];   // mask words, dbl-buf
    const int lane = threadIdx.x & 31;
    const int wave = threadIdx.x >> 5;
    const int ln   = lane & 15;
    const int half = lane >> 4;
    const int wg   = blockIdx.x * 8 + wave;
    const int qt   = wg % (SEQ / 16);
    const int bh   = wg / (SEQ / 16);
    const int h    = bh % NH;
    const int b    = bh / NH;

    // Q A-fragments (two K-slices of 32 over head dim 64), loop-resident.
    AFrag qa[2];
    const __bf16* qrow = qb + (size_t)(b * SEQ + qt * 16 + ln) * D_MODEL + h * DH;
#pragma unroll
    for (int ks = 0; ks < 2; ++ks) {
        ld8_bf16(qrow + ks * 32 + half * 8,      qa[ks].h);
        ld8_bf16(qrow + ks * 32 + 16 + half * 8, qa[ks].h + 8);
    }

    const v8f z = {0.f, 0.f, 0.f, 0.f, 0.f, 0.f, 0.f, 0.f};
    v8f o[4];
    o[0] = z; o[1] = z; o[2] = z; o[3] = z;
    float mx[8], ls[8];
#pragma unroll
    for (int r = 0; r < 8; ++r) { mx[r] = NEG_BIG; ls[r] = 0.f; }

    const __bf16*   kbase  = kb  + (size_t)b * SEQ * D_MODEL + h * DH;
    const __bf16*   vbase  = vtb + (size_t)(b * D_MODEL + h * DH) * SEQ;
    const uint32_t* mwbase = mw + ((size_t)b * SEQ + qt * 16) * MWPR;
    const float scale = 0.125f;   // 1/sqrt(64)

    // Async copy of this kv-tile's 16 mask words into LDS (lanes 0..15).
    auto issue_mask = [&](int kv0, int buf) {
        if (lane < 16) {
            const uint32_t* src = mwbase + (size_t)lane * MWPR + (kv0 >> 5);
            uint32_t dst = (uint32_t)(uintptr_t)&mlds[wave][buf][lane];
            asm volatile("global_load_async_to_lds_b32 %0, %1, off"
                         :: "v"(dst), "v"(src) : "memory");
        }
    };

    issue_mask(0, 0);
    int buf = 0;
    for (int kv0 = 0; kv0 < SEQ; kv0 += 32) {
        // Current tile's mask words are in LDS; prefetch the next tile.
        asm volatile("s_wait_asynccnt 0x0" ::: "memory");
        if (kv0 + 32 < SEQ) issue_mask(kv0 + 32, buf ^ 1);

        // --- scores S[16 x 32] = Q[16 x 64] * K^T[64 x 32] ---
        v8f c[2];
        c[0] = z; c[1] = z;
#pragma unroll
        for (int nt = 0; nt < 2; ++nt) {
            const __bf16* krow = kbase + (size_t)(kv0 + nt * 16 + ln) * D_MODEL;
#pragma unroll
            for (int ks = 0; ks < 2; ++ks) {
                AFrag kf;
                ld8_bf16(krow + ks * 32 + half * 8,      kf.h);
                ld8_bf16(krow + ks * 32 + 16 + half * 8, kf.h + 8);
                c[nt] = __builtin_amdgcn_wmma_f32_16x16x32_bf16(
                    false, qa[ks].v, false, kf.v, (short)0, c[nt], false, false);
            }
        }
        // --- online softmax over this kv tile ---
#pragma unroll
        for (int r = 0; r < 8; ++r) {
            const int mrow = half * 8 + r;          // row within 16-query tile
            const uint32_t wr = mlds[wave][buf][mrow];
            float s0 = c[0][r] * scale;
            float s1 = c[1][r] * scale;
            if (((wr >> ln) & 1u) == 0u)        s0 = NEG_BIG;
            if (((wr >> (16 + ln)) & 1u) == 0u) s1 = NEG_BIG;
            float rm = fmaxf(s0, s1);
            rm = fmaxf(rm, __shfl_xor(rm, 1));
            rm = fmaxf(rm, __shfl_xor(rm, 2));
            rm = fmaxf(rm, __shfl_xor(rm, 4));
            rm = fmaxf(rm, __shfl_xor(rm, 8));
            const float mn   = fmaxf(mx[r], rm);
            const float corr = __expf(mx[r] - mn);
            const float p0   = __expf(s0 - mn);
            const float p1   = __expf(s1 - mn);
            float rs = p0 + p1;
            rs += __shfl_xor(rs, 1);
            rs += __shfl_xor(rs, 2);
            rs += __shfl_xor(rs, 4);
            rs += __shfl_xor(rs, 8);
            ls[r] = ls[r] * corr + rs;
            mx[r] = mn;
#pragma unroll
            for (int nt = 0; nt < 4; ++nt) o[nt][r] *= corr;
            plds[wave][mrow][ln]      = (__bf16)p0;
            plds[wave][mrow][16 + ln] = (__bf16)p1;
        }
        // --- re-gather P into A-fragment layout (same-wave LDS) ---
        AFrag pf;
        ld8_bf16(&plds[wave][ln][half * 8],      pf.h);
        ld8_bf16(&plds[wave][ln][16 + half * 8], pf.h + 8);
        // --- O[16 x 64] += P[16 x 32] * V[32 x 64] (V^T: contiguous) ---
#pragma unroll
        for (int nt = 0; nt < 4; ++nt) {
            const __bf16* vrow = vbase + (size_t)(nt * 16 + ln) * SEQ + kv0;
            AFrag vf;
            ld8_bf16(vrow + half * 8,      vf.h);
            ld8_bf16(vrow + 16 + half * 8, vf.h + 8);
            o[nt] = __builtin_amdgcn_wmma_f32_16x16x32_bf16(
                false, pf.v, false, vf.v, (short)0, o[nt], false, false);
        }
        buf ^= 1;
    }
    // --- normalize and store O as bf16 [B, S, H*64] (concat-heads) ---
#pragma unroll
    for (int nt = 0; nt < 4; ++nt) {
#pragma unroll
        for (int r = 0; r < 8; ++r) {
            const int s    = qt * 16 + half * 8 + r;
            const float iv = 1.0f / ls[r];
            ob[(size_t)(b * SEQ + s) * D_MODEL + h * DH + nt * 16 + ln] =
                (__bf16)(o[nt][r] * iv);
        }
    }
}

extern "C" void kernel_launch(void* const* d_in, const int* in_sizes, int n_in,
                              void* d_out, int out_size, void* d_ws, size_t ws_size,
                              hipStream_t stream) {
    const float*   qx   = (const float*)d_in[0];
    const float*   kvx  = (const float*)d_in[1];
    const uint8_t* mask = (const uint8_t*)d_in[2];
    const float*   Wq   = (const float*)d_in[3];
    const float*   bq   = (const float*)d_in[4];
    const float*   Wk   = (const float*)d_in[5];
    const float*   bk   = (const float*)d_in[6];
    const float*   Wv   = (const float*)d_in[7];
    const float*   bv   = (const float*)d_in[8];
    const float*   Wo   = (const float*)d_in[9];
    const float*   bo   = (const float*)d_in[10];
    float* out = (float*)d_out;

    const size_t NTOK  = (size_t)BATCH * SEQ;      // 4096 rows
    const size_t ELEMS = NTOK * D_MODEL;           // 4M  (act tensors)
    const size_t WELEM = (size_t)D_MODEL * D_MODEL;// 1M  (weight tensors)

    __bf16* qb   = (__bf16*)d_ws;                  // Q   bf16 [B,S,1024]
    __bf16* kb   = qb  + ELEMS;                    // K   bf16 [B,S,1024]
    __bf16* vtb  = kb  + ELEMS;                    // V^T bf16 [B,1024,S]
    __bf16* ab   = vtb + ELEMS;                    // attn out bf16 [B,S,1024]
    __bf16* qxb  = ab  + ELEMS;                    // bf16 inputs
    __bf16* kvxb = qxb + ELEMS;
    __bf16* wqb  = kvxb + ELEMS;                   // bf16 weights
    __bf16* wkb  = wqb + WELEM;
    __bf16* wvb  = wkb + WELEM;
    __bf16* wob  = wvb + WELEM;
    uint32_t* mwp = (uint32_t*)(wob + WELEM);      // packed mask bits (1MB)

    dim3 blk(256);

    // 1) bf16 conversions (one-time; amortized over tile reuse).
    cvt_bf16<<<dim3((unsigned)(ELEMS / 8 / 256)), blk, 0, stream>>>(qx,  qxb,  (int)(ELEMS / 8));
    cvt_bf16<<<dim3((unsigned)(ELEMS / 8 / 256)), blk, 0, stream>>>(kvx, kvxb, (int)(ELEMS / 8));
    cvt_bf16<<<dim3((unsigned)(WELEM / 8 / 256)), blk, 0, stream>>>(Wq, wqb, (int)(WELEM / 8));
    cvt_bf16<<<dim3((unsigned)(WELEM / 8 / 256)), blk, 0, stream>>>(Wk, wkb, (int)(WELEM / 8));
    cvt_bf16<<<dim3((unsigned)(WELEM / 8 / 256)), blk, 0, stream>>>(Wv, wvb, (int)(WELEM / 8));
    cvt_bf16<<<dim3((unsigned)(WELEM / 8 / 256)), blk, 0, stream>>>(Wo, wob, (int)(WELEM / 8));

    // 2) bit-pack the attention mask.
    const int mbytes = BATCH * SEQ * SEQ;
    maskpack<<<dim3((unsigned)(mbytes / 256)), blk, 0, stream>>>(mask, mwp, mbytes);

    // 3) fused per-head projections as 4096x1024x1024 GEMMs.
    dim3 ggrid((unsigned)((NTOK / 64) * (D_MODEL / 64) / 8));   // 128 blocks
    gemm64<0><<<ggrid, blk, 0, stream>>>(qxb,  wqb, bq, qb,  (int)NTOK);
    gemm64<0><<<ggrid, blk, 0, stream>>>(kvxb, wkb, bk, kb,  (int)NTOK);
    gemm64<1><<<ggrid, blk, 0, stream>>>(kvxb, wvb, bv, vtb, (int)NTOK);

    // 4) flash attention.
    dim3 agrid((unsigned)((BATCH * NH * (SEQ / 16)) / 8));      // 512 blocks
    attn16<<<agrid, blk, 0, stream>>>(qb, kb, vtb, mwp, ab);

    // 5) output projection straight into d_out (fp32 + bias).
    gemm64<2><<<ggrid, blk, 0, stream>>>(ab, wob, bo, out, (int)NTOK);
}
